// MXLinear_66889820668548
// MI455X (gfx1250) — matrix operand order repaired and checked by reading, not compiled, not run on hardware
//
#include <hip/hip_runtime.h>

typedef __attribute__((ext_vector_type(16))) _Float16 v16h;
typedef __attribute__((ext_vector_type(8)))  _Float16 v8h;
typedef __attribute__((ext_vector_type(2)))  _Float16 h2;
typedef __attribute__((ext_vector_type(8)))  float    v8f;
typedef unsigned __attribute__((ext_vector_type(4))) u32x4;
typedef int      __attribute__((ext_vector_type(8))) i32x8;
typedef int      __attribute__((ext_vector_type(4))) i32x4;

#define M_TOT 8192
#define K_TOT 4096
#define N_TOT 11008
#define BK    32
#define NKC   (K_TOT / BK)   // 128 K-chunks
#define STRA  48             // A LDS row stride (f16): 32 + 16 pad, 16B aligned rows
#define STRB  136            // B LDS row stride (f16): 128 + 8 pad, 16B aligned rows
#define SPST  17             // packed staging row stride in dwords (16 + 1 pad via TDM)

// ---- Tensor Data Mover: 2-D tile load, D# per CDNA5 ISA ch.8 (6-arg clang-23 form) ----
__device__ __forceinline__ void tdm_load_2d(unsigned lds_off, const void* gptr,
                                            unsigned tile0, unsigned tile1,
                                            unsigned tdim0, unsigned tdim1,
                                            unsigned long long stride0,
                                            bool pad, unsigned pad_interval,
                                            unsigned pad_amount) {
    unsigned long long ga = (unsigned long long)(uintptr_t)gptr;
    u32x4 g0;
    g0[0] = 1u;                                      // count=1 (valid), user mode
    g0[1] = lds_off;                                 // lds_addr [63:32]
    g0[2] = (unsigned)(ga & 0xFFFFFFFFull);          // global_addr [95:64]
    g0[3] = (unsigned)((ga >> 32) & 0x1FFFFFFull)    // global_addr [120:96]
          | 0x80000000u;                             // type=2 ("image") [127:126]
    i32x8 g1;
    unsigned d0 = (2u << 16);                        // data_size = 4B
    if (pad) d0 |= (1u << 20) | (pad_interval << 22) | (pad_amount << 25);
    g1[0] = (int)d0;                                 // wg_mask=0 (not in cluster)
    g1[1] = (int)((tdim0 & 0xFFFFu) << 16);          // tensor_dim0[15:0] @ [63:48]
    g1[2] = (int)((tdim0 >> 16) | ((tdim1 & 0xFFFFu) << 16));
    g1[3] = (int)((tdim1 >> 16) | (tile0 << 16));    // tile_dim0 @ [127:112]
    g1[4] = (int)(tile1 & 0xFFFFu);                  // tile_dim1, tile_dim2=0
    g1[5] = (int)(unsigned)(stride0 & 0xFFFFFFFFull);
    g1[6] = (int)(unsigned)((stride0 >> 32) & 0xFFFFull);  // stride1=0 (unused, 2D)
    g1[7] = 0;
    i32x4 z4 = {};
    i32x8 z8 = {};
    __builtin_amdgcn_tensor_load_to_lds(g0, g1, z4, z4, z8, 0);
}

__global__ void __launch_bounds__(256)
mxfp4_linear_wmma(const float* __restrict__ A,      // [8192, 4096] fp32 activations
                  const float* __restrict__ bias,   // [11008]
                  const int*   __restrict__ packed, // [11008, 2048] two FP4 codes / dword
                  const int*   __restrict__ scales, // [11008, 128] E8M0 exponents
                  float*       __restrict__ out)    // [8192, 11008]
{
    __shared__ __align__(16) _Float16 As[128 * STRA];   // 12 KB   f16 activations
    __shared__ __align__(16) _Float16 Bs[BK * STRB];    // 8.5 KB  f16 dequant weights
    __shared__ __align__(16) int Sp[2][128 * SPST];     // 17 KB   TDM packed staging (x2)
    __shared__ __align__(16) int Ss[2][128];            // 1 KB    TDM scale staging (x2)

    const int t    = threadIdx.x;
    const int lane = t & 31;
    const int wave = t >> 5;          // 8 waves
    const int wm   = wave >> 2;       // 0..1 : wave M block (64 rows)
    const int wn   = wave & 3;        // 0..3 : wave N block (32 cols)
    const int h    = lane >> 4;
    const int ml   = lane & 15;

    const int tileN = blockIdx.x * 128;
    const int tileM = blockIdx.y * 128;

    // A loader: thread -> (row 0..127, which 16-float half of the 32-wide chunk)
    const int a_row  = t >> 1;
    const int a_colh = (t & 1) * 16;
    const float* a_ptr = A + (size_t)(tileM + a_row) * K_TOT + a_colh;

    // B dequant: thread -> k-pair q (rows 2q,2q+1) for 8 consecutive n columns
    const int b_q = t >> 4;            // 0..15
    const int b_g = (t & 15) * 8;      // n-local base

    const int* packedBase = packed + (size_t)tileN * (K_TOT / 2);
    const int* scalesBase = scales + (size_t)tileN * NKC;

    const float bias0 = bias[tileN + wn * 32 + 0  + ml];
    const float bias1 = bias[tileN + wn * 32 + 16 + ml];

    v8f zero = {};
    v8f acc[4][2];
#pragma unroll
    for (int mf = 0; mf < 4; ++mf)
#pragma unroll
        for (int nf = 0; nf < 2; ++nf) acc[mf][nf] = zero;

    float4 aReg[4];

    auto issueTDM = [&](int kc, int buf) {
        // packed weights: 128 rows x 16 dwords, row stride 2048 dwords,
        // TDM pads +1 dword per 16 -> staging row stride 17 dwords
        tdm_load_2d((unsigned)(uintptr_t)&Sp[buf][0],
                    packedBase + kc * (BK / 2),
                    16, 128, 16, 128, (unsigned long long)(K_TOT / 2),
                    true, 3 /*16 DWORDs*/, 0 /*+1 DWORD*/);
        // E8M0 scales: 128 rows x 1 dword, row stride 128 dwords
        tdm_load_2d((unsigned)(uintptr_t)&Ss[buf][0],
                    scalesBase + kc,
                    1, 128, 1, 128, (unsigned long long)NKC,
                    false, 0, 0);
    };

    auto loadA = [&](int kc) {
        const float4* ap = (const float4*)(a_ptr + kc * BK);
#pragma unroll
        for (int j = 0; j < 4; ++j) aReg[j] = ap[j];
    };

    auto storeA = [&]() {
        _Float16 tmp[16];
#pragma unroll
        for (int j = 0; j < 4; ++j) {
            tmp[4 * j + 0] = (_Float16)aReg[j].x;
            tmp[4 * j + 1] = (_Float16)aReg[j].y;
            tmp[4 * j + 2] = (_Float16)aReg[j].z;
            tmp[4 * j + 3] = (_Float16)aReg[j].w;
        }
        v8h alo, ahi;
#pragma unroll
        for (int e = 0; e < 8; ++e) { alo[e] = tmp[e]; ahi[e] = tmp[8 + e]; }
        *(v8h*)&As[a_row * STRA + a_colh]     = alo;
        *(v8h*)&As[a_row * STRA + a_colh + 8] = ahi;
    };

    // FP4xE8M0 -> f16 via v_perm byte-LUT; f16 magnitude bits = lut[m] << 9
    auto dequant = [&](int buf) {
        int pv[8];
#pragma unroll
        for (int i = 0; i < 8; ++i)
            pv[i] = Sp[buf][(b_g + i) * SPST + b_q];
        int4 sA = *(const int4*)&Ss[buf][b_g];
        int4 sB = *(const int4*)&Ss[buf][b_g + 4];
        int se[8] = {sA.x, sA.y, sA.z, sA.w, sB.x, sB.y, sB.z, sB.w};
        unsigned r[8];
#pragma unroll
        for (int i = 0; i < 8; ++i) {
            unsigned v   = (unsigned)pv[i];
            unsigned sel = (v & 7u) | ((v & 0x70u) << 4);          // both nibble mags
            unsigned p   = __builtin_amdgcn_perm(0x23222120u, 0x1F1E1C00u, sel);
            unsigned f   = ((p & 0xFFu) << 9) | ((p & 0xFF00u) << 17)  // magnitudes
                         | ((v & 8u) << 12) | ((v & 0x80u) << 24);     // signs
            unsigned sf  = ((unsigned)(se[i] - 112)) << 10;            // 2^(E-127) in f16
            unsigned s2  = sf | (sf << 16);
            h2 prod = __builtin_bit_cast(h2, f) * __builtin_bit_cast(h2, s2);
            r[i] = __builtin_bit_cast(unsigned, prod);
        }
        uint4 ev, od;   // de-interleave (even k, odd k) rows with v_perm
        ev.x = __builtin_amdgcn_perm(r[1], r[0], 0x05040100u);
        ev.y = __builtin_amdgcn_perm(r[3], r[2], 0x05040100u);
        ev.z = __builtin_amdgcn_perm(r[5], r[4], 0x05040100u);
        ev.w = __builtin_amdgcn_perm(r[7], r[6], 0x05040100u);
        od.x = __builtin_amdgcn_perm(r[1], r[0], 0x07060302u);
        od.y = __builtin_amdgcn_perm(r[3], r[2], 0x07060302u);
        od.z = __builtin_amdgcn_perm(r[5], r[4], 0x07060302u);
        od.w = __builtin_amdgcn_perm(r[7], r[6], 0x07060302u);
        *(uint4*)&Bs[(2 * b_q)     * STRB + b_g] = ev;
        *(uint4*)&Bs[(2 * b_q + 1) * STRB + b_g] = od;
    };

    auto compute = [&]() {
        v16h bfrag[2];
#pragma unroll
        for (int nf = 0; nf < 2; ++nf) {
            const _Float16* p = &Bs[wn * 32 + nf * 16 + ml];
#pragma unroll
            for (int s = 0; s < 16; ++s) {
                const int k = (s < 8) ? (8 * h + s) : (16 + 8 * h + (s - 8));
                bfrag[nf][s] = p[k * STRB];
            }
        }
#pragma unroll
        for (int mf = 0; mf < 4; ++mf) {
            const _Float16* p = &As[(wm * 64 + mf * 16 + ml) * STRA + h * 8];
            v8h a0 = *(const v8h*)p;
            v8h a1 = *(const v8h*)(p + 16);
            v16h afrag = __builtin_shufflevector(a0, a1, 0, 1, 2, 3, 4, 5, 6, 7,
                                                 8, 9, 10, 11, 12, 13, 14, 15);
#pragma unroll
            for (int nf = 0; nf < 2; ++nf) {
                acc[mf][nf] = __builtin_amdgcn_wmma_f32_16x16x32_f16(
                    false, afrag, false, bfrag[nf],
                    (short)0, acc[mf][nf], false, false);
            }
        }
    };

    // ---- pipelined K loop: TDM double-buffers weights; A prefetched in regs ----
    if (wave == 0) issueTDM(0, 0);
    loadA(0);
    for (int kc = 0; kc < NKC; ++kc) {
        const int buf = kc & 1;
        __syncthreads();                         // prior compute done: As/Bs/staging free
        storeA();
        if (kc + 1 < NKC) {
            if (wave == 0) issueTDM(kc + 1, buf ^ 1);   // next chunk's DMA in flight
            loadA(kc + 1);                               // overlaps WMMA below
            if (wave == 0) __builtin_amdgcn_s_wait_tensorcnt(2);  // this chunk's 2 done
        } else {
            if (wave == 0) __builtin_amdgcn_s_wait_tensorcnt(0);
        }
        __syncthreads();                         // As + staging[buf] visible
        dequant(buf);
        __syncthreads();                         // Bs ready
        compute();
    }

    // ---- epilogue: C frag VGPR r -> M = base + r + 8h, lane&15 -> N ----
#pragma unroll
    for (int mf = 0; mf < 4; ++mf) {
        const int mbase = tileM + wm * 64 + mf * 16 + 8 * h;
#pragma unroll
        for (int nf = 0; nf < 2; ++nf) {
            const int n   = tileN + wn * 32 + nf * 16 + ml;
            const float bv = nf ? bias1 : bias0;
#pragma unroll
            for (int r = 0; r < 8; ++r) {
                out[(size_t)(mbase + r) * N_TOT + n] = acc[mf][nf][r] + bv;
            }
        }
    }
}

extern "C" void kernel_launch(void* const* d_in, const int* in_sizes, int n_in,
                              void* d_out, int out_size, void* d_ws, size_t ws_size,
                              hipStream_t stream) {
    const float* input  = (const float*)d_in[0];  // [4,2048,4096] fp32
    const float* bias   = (const float*)d_in[1];  // [11008] fp32
    const int*   packed = (const int*)d_in[2];    // [11008,2048] int32
    const int*   scales = (const int*)d_in[3];    // [11008,128] int32
    float* out = (float*)d_out;                   // [4,2048,11008] fp32

    dim3 grid(N_TOT / 128, M_TOT / 128);          // 86 x 64 workgroups
    mxfp4_linear_wmma<<<grid, dim3(256), 0, stream>>>(input, bias, packed, scales, out);
}
